// DFC_KL_17523466567753
// MI455X (gfx1250) — matrix operand order repaired and checked by reading, not compiled, not run on hardware
//
#include <hip/hip_runtime.h>
#include <hip/hip_bf16.h>
#include <math.h>

// ---------------------------------------------------------------------------
// Problem constants (from reference): x is (768, 256, 256) f32.
//   N = 65536 pixels, C = 768 channels, K = 64 clusters, 10 stages.
// ---------------------------------------------------------------------------
#define NPIX   65536
#define NCH    768
#define NCLU   64
#define STAGES 10
#define KSTEPS 24      // 768 / 32 (WMMA K per step)
#define MTILES 4       // 64 / 16 cluster tiles

typedef __attribute__((ext_vector_type(16))) __bf16 v16bf;
typedef __attribute__((ext_vector_type(8)))  float  v8f;

// ---------------------------------------------------------------------------
// Kernel 1 (once): logx in bf16, pixel-major [N][C], via LDS-tiled transpose.
// Input x is channel-major [C][N].
// ---------------------------------------------------------------------------
__global__ void k_logx(const float* __restrict__ x, __bf16* __restrict__ logxB) {
    __shared__ float tile[32][33];
    int n0 = blockIdx.x * 32;
    int c0 = blockIdx.y * 32;
    int tx = threadIdx.x;          // 0..31
    int ty = threadIdx.y;          // 0..7
#pragma unroll
    for (int i = 0; i < 32; i += 8) {
        float v = x[(size_t)(c0 + ty + i) * NPIX + (n0 + tx)];
        if (v == 0.0f) v = 1e-6f;
        tile[ty + i][tx] = __logf(v);
    }
    __syncthreads();
#pragma unroll
    for (int i = 0; i < 32; i += 8) {
        logxB[(size_t)(n0 + ty + i) * NCH + (c0 + tx)] = (__bf16)tile[tx][ty + i];
    }
}

// ---------------------------------------------------------------------------
// Kernel 2 (once): seed prototypes from the 8x8 grid pixels; counts := 1
// (so stage-0 "present" mask is all-true).
// ---------------------------------------------------------------------------
__global__ void k_init(const float* __restrict__ x, float* __restrict__ proto,
                       float* __restrict__ counts) {
    int t = blockIdx.x * blockDim.x + threadIdx.x;
    if (t >= NCLU * NCH) return;
    int k = t / NCH, c = t % NCH;
    int n = (k >> 3) * 32 * 256 + (k & 7) * 32;      // grid seed pixel index
    float v = x[(size_t)c * NPIX + n];
    if (v == 0.0f) v = 1e-6f;
    proto[t] = v;
    if (t < NCLU) counts[t] = 1.0f;
}

// ---------------------------------------------------------------------------
// Per stage: ent[k] = sum_c p*log(p)  (or +inf for dead clusters so that
// dis = ent - dot == +inf and the cluster can never win the argmin).
// One wave per cluster.
// ---------------------------------------------------------------------------
__global__ void k_ent(const float* __restrict__ proto, const float* __restrict__ counts,
                      float* __restrict__ ent) {
    int k = blockIdx.x;
    int lane = threadIdx.x;        // 0..31
    float s = 0.0f;
#pragma unroll
    for (int i = 0; i < NCH / 32; ++i) {
        float p = proto[(size_t)k * NCH + i * 32 + lane];
        s += p * __logf(p);
    }
#pragma unroll
    for (int off = 16; off > 0; off >>= 1) s += __shfl_down(s, off);
    if (lane == 0) ent[k] = (counts[k] > 0.0f) ? s : __builtin_inff();
}

// ---------------------------------------------------------------------------
// Per stage: pack proto (f32 [64][768]) into WMMA A-fragment layout (bf16).
// Fragment f = mtile*24 + kstep holds 32 lanes x 16 bf16 contiguous.
// A 16x32 k order used here (contiguous): lanes 0-15 = rows, K 0..15; lanes 16-31
// same rows, K 16..31 (element j -> local K).
// ---------------------------------------------------------------------------
__global__ void k_pack(const float* __restrict__ proto, __bf16* __restrict__ protoA) {
    int t = blockIdx.x * blockDim.x + threadIdx.x;
    if (t >= NCLU * NCH) return;
    int f    = t >> 9;             // fragment index (512 elems each)
    int rem  = t & 511;
    int lane = rem >> 4;
    int j    = rem & 15;           // element within lane's v16bf
    int v = j >> 1, h = j & 1;
    int mtile = f / KSTEPS, kstep = f % KSTEPS;
    int row = lane & 15;
    int kl  = j + ((lane >= 16) ? 16 : 0);
    int m = mtile * 16 + row;
    int k = kstep * 32 + kl;
    protoA[t] = (__bf16)proto[(size_t)m * NCH + k];
}

// ---------------------------------------------------------------------------
// Per stage: assignment. One wave = 16 pixels x all 64 clusters.
// dot = proto @ logx.T via v_wmma_f32_16x16x32_bf16 (4 accs x 24 k-steps),
// then dis[k] = ent[k] - dot[k], argmin over k via per-wave LDS tile.
// logxB in L2 (96 MB < 192 MB) -> this kernel runs at L2 bandwidth.
// ---------------------------------------------------------------------------
__global__ void __launch_bounds__(256)
k_assign(const __bf16* __restrict__ logxB, const __bf16* __restrict__ protoA,
         const float* __restrict__ ent, unsigned char* __restrict__ cluster8) {
    __shared__ float entS[NCLU];
    __shared__ float disS[8][NCLU * 16];     // per-wave 64x16 distance tile (32 KB)

    int tid = threadIdx.x;
    if (tid < NCLU) entS[tid] = ent[tid];
    __syncthreads();

    int wave = tid >> 5;
    int lane = tid & 31;
    int ntile = blockIdx.x * 8 + wave;
    int n0 = ntile * 16;

    // B fragment: lane = pixel column (n0 + lane&15); lanes 0-15 take K 0..15
    // of each 32-chunk, lanes 16-31 take K 16..31. 16 contiguous bf16 per lane.
    const __bf16* brow = logxB + (size_t)(n0 + (lane & 15)) * NCH + ((lane >> 4) << 4);
    const __bf16* arow = protoA + (size_t)lane * 16;

    v8f acc0 = {}, acc1 = {}, acc2 = {}, acc3 = {};
    for (int s = 0; s < KSTEPS; ++s) {
        v16bf b  = *(const v16bf*)(brow + s * 32);
        v16bf a0 = *(const v16bf*)(arow + (size_t)((0 * KSTEPS + s) * 512));
        v16bf a1 = *(const v16bf*)(arow + (size_t)((1 * KSTEPS + s) * 512));
        v16bf a2 = *(const v16bf*)(arow + (size_t)((2 * KSTEPS + s) * 512));
        v16bf a3 = *(const v16bf*)(arow + (size_t)((3 * KSTEPS + s) * 512));
        acc0 = __builtin_amdgcn_wmma_f32_16x16x32_bf16(false, a0, false, b, (short)0, acc0, false, false);
        acc1 = __builtin_amdgcn_wmma_f32_16x16x32_bf16(false, a1, false, b, (short)0, acc1, false, false);
        acc2 = __builtin_amdgcn_wmma_f32_16x16x32_bf16(false, a2, false, b, (short)0, acc2, false, false);
        acc3 = __builtin_amdgcn_wmma_f32_16x16x32_bf16(false, a3, false, b, (short)0, acc3, false, false);
    }

    // C/D layout: lane 0-15 hold rows r (VGPR r), lanes 16-31 rows r+8; col = lane&15.
    float* myDis = &disS[wave][0];
    int rbase = (lane >> 4) << 3;
    int col = lane & 15;
#pragma unroll
    for (int r = 0; r < 8; ++r) {
        myDis[(0  + rbase + r) * 16 + col] = acc0[r];
        myDis[(16 + rbase + r) * 16 + col] = acc1[r];
        myDis[(32 + rbase + r) * 16 + col] = acc2[r];
        myDis[(48 + rbase + r) * 16 + col] = acc3[r];
    }
    __syncthreads();

    if (lane < 16) {
        float best = __builtin_inff();
        int bk = 0;
#pragma unroll 4
        for (int k = 0; k < NCLU; ++k) {
            float d = entS[k] - myDis[k * 16 + lane];   // kl distance (first-min wins)
            if (d < best) { best = d; bk = k; }
        }
        cluster8[n0 + lane] = (unsigned char)bk;
    }
}

// ---------------------------------------------------------------------------
// Per stage: zero sums + counts.
// ---------------------------------------------------------------------------
__global__ void k_zero(float* __restrict__ sums, float* __restrict__ counts) {
    int t = blockIdx.x * blockDim.x + threadIdx.x;
    if (t < NCLU * NCH) sums[t] = 0.0f;
    if (t < NCLU) counts[t] = 0.0f;
}

// ---------------------------------------------------------------------------
// Per stage: counts via LDS bucket reduce (ds_add_f32) + 64 global atomics/block.
// ---------------------------------------------------------------------------
__global__ void k_counts(const unsigned char* __restrict__ cl, float* __restrict__ counts) {
    __shared__ float s[NCLU];
    int tid = threadIdx.x;
    if (tid < NCLU) s[tid] = 0.0f;
    __syncthreads();
    int n = blockIdx.x * 256 + tid;
    atomicAdd(&s[cl[n]], 1.0f);
    __syncthreads();
    if (tid < NCLU) atomicAdd(&counts[tid], s[tid]);
}

// ---------------------------------------------------------------------------
// Per stage: per-cluster channel sums. Channel-major so x reads are coalesced;
// LDS buckets per (channel, pixel-chunk) block, then 64 global atomics.
// ---------------------------------------------------------------------------
__global__ void k_sums(const float* __restrict__ x, const unsigned char* __restrict__ cl,
                       float* __restrict__ sums) {
    __shared__ float s[NCLU];
    int c = blockIdx.x;
    int tid = threadIdx.x;
    if (tid < NCLU) s[tid] = 0.0f;
    __syncthreads();
    const float* xc = x + (size_t)c * NPIX;
    int base = blockIdx.y * 4096;
#pragma unroll 4
    for (int i = 0; i < 16; ++i) {
        int n = base + i * 256 + tid;
        float v = xc[n];
        if (v == 0.0f) v = 1e-6f;
        atomicAdd(&s[cl[n]], v);
    }
    __syncthreads();
    if (tid < NCLU) atomicAdd(&sums[(size_t)tid * NCH + c], s[tid]);
}

// ---------------------------------------------------------------------------
// Per stage: proto = present ? sums/counts : 1.0
// ---------------------------------------------------------------------------
__global__ void k_update(const float* __restrict__ sums, const float* __restrict__ counts,
                         float* __restrict__ proto) {
    int t = blockIdx.x * blockDim.x + threadIdx.x;
    if (t >= NCLU * NCH) return;
    float cnt = counts[t / NCH];
    proto[t] = (cnt > 0.0f) ? sums[t] / cnt : 1.0f;
}

// ---------------------------------------------------------------------------
// Final: compact-label scan (tiny, serial) then relabel + num_clusters.
// ---------------------------------------------------------------------------
__global__ void k_scan(const float* __restrict__ counts, int* __restrict__ newid,
                       float* __restrict__ numc) {
    if (threadIdx.x == 0) {
        int run = 0;
        for (int k = 0; k < NCLU; ++k) {
            run += (counts[k] > 0.0f) ? 1 : 0;
            newid[k] = run - 1;
        }
        numc[0] = (float)run;
    }
}

__global__ void k_out(const unsigned char* __restrict__ cl, const int* __restrict__ newid,
                      const float* __restrict__ numc, float* __restrict__ out, int out_size) {
    int t = blockIdx.x * blockDim.x + threadIdx.x;
    if (t < NPIX && t < out_size) out[t] = (float)newid[cl[t]];
    if (t == 0 && out_size > NPIX) out[NPIX] = numc[0];
}

// ---------------------------------------------------------------------------
// Host launcher: fixed, deterministic launch sequence (graph-capture safe).
// ---------------------------------------------------------------------------
extern "C" void kernel_launch(void* const* d_in, const int* in_sizes, int n_in,
                              void* d_out, int out_size, void* d_ws, size_t ws_size,
                              hipStream_t stream) {
    (void)in_sizes; (void)n_in; (void)ws_size;
    const float* x = (const float*)d_in[0];
    float* out = (float*)d_out;

    // Workspace layout (256B aligned slices). Total ~96.6 MB.
    char* w = (char*)d_ws;
    auto take = [&](size_t bytes) -> char* {
        char* p = w;
        w += (bytes + 255) & ~(size_t)255;
        return p;
    };
    __bf16*        logxB  = (__bf16*)take((size_t)NPIX * NCH * sizeof(__bf16));   // 96 MB, L2-resident
    __bf16*        protoA = (__bf16*)take((size_t)NCLU * NCH * sizeof(__bf16));   // A fragments
    float*         proto  = (float*) take((size_t)NCLU * NCH * sizeof(float));
    float*         sums   = (float*) take((size_t)NCLU * NCH * sizeof(float));
    float*         ent    = (float*) take(NCLU * sizeof(float));
    float*         counts = (float*) take(NCLU * sizeof(float));
    float*         numc   = (float*) take(sizeof(float));
    int*           newid  = (int*)   take(NCLU * sizeof(int));
    unsigned char* cl     = (unsigned char*)take(NPIX);

    // One-time prep: bf16 logx (transposed to pixel-major) + seed prototypes.
    k_logx<<<dim3(NPIX / 32, NCH / 32), dim3(32, 8), 0, stream>>>(x, logxB);
    k_init<<<(NCLU * NCH + 255) / 256, 256, 0, stream>>>(x, proto, counts);

    for (int s = 0; s < STAGES; ++s) {
        k_ent   <<<NCLU, 32, 0, stream>>>(proto, counts, ent);
        k_pack  <<<(NCLU * NCH + 255) / 256, 256, 0, stream>>>(proto, protoA);
        k_assign<<<NPIX / (16 * 8), 256, 0, stream>>>(logxB, protoA, ent, cl);
        k_zero  <<<(NCLU * NCH + 255) / 256, 256, 0, stream>>>(sums, counts);
        k_counts<<<NPIX / 256, 256, 0, stream>>>(cl, counts);
        k_sums  <<<dim3(NCH, NPIX / 4096), 256, 0, stream>>>(x, cl, sums);
        k_update<<<(NCLU * NCH + 255) / 256, 256, 0, stream>>>(sums, counts, proto);
    }

    k_scan<<<1, 32, 0, stream>>>(counts, newid, numc);
    k_out <<<NPIX / 256, 256, 0, stream>>>(cl, newid, numc, out, out_size);
}
